// Head_85212151153417
// MI455X (gfx1250) — matrix operand order repaired and checked by reading, not compiled, not run on hardware
//
#include <hip/hip_runtime.h>

// ---------------------------------------------------------------------------
// Fused attention head for MI455X (gfx1250, wave32, WMMA bf16 16x16x32).
//   B=8, T=2048, C=2048, H=128
// Phase 0: weights -> bf16, transposed wT[mat][h][c]
// Phase 1: QKV projection + RoPE; one wave = 16x128 tile (A-frag reused x8)
// Phase 2: flash-attention; 8 waves/WG split keys; K blocks streamed into LDS
//          with GLOBAL_LOAD_ASYNC_TO_LDS_B128 (double buffered, ASYNCcnt);
//          LDS log-sum-exp merge across waves.
// ---------------------------------------------------------------------------

typedef __attribute__((ext_vector_type(16))) __bf16 v16bf;
typedef __attribute__((ext_vector_type(8)))  float  v8f;

#define B_  8
#define T_  2048
#define C_  2048
#define H_  128

__device__ __forceinline__ __bf16 bf(float f) { return (__bf16)f; }

__device__ __forceinline__ v8f wmma_bf16(v16bf a, v16bf b, v8f c) {
  return __builtin_amdgcn_wmma_f32_16x16x32_bf16(false, a, false, b,
                                                 (short)0, c, false, false);
}

// ---------------------------------------------------------------------------
// Phase 0: convert + transpose weights: wT[mat][h][c] = bf16(w[c][h])
// ---------------------------------------------------------------------------
__global__ __launch_bounds__(256)
void wconv_kernel(const float* __restrict__ wq,
                  const float* __restrict__ wk,
                  const float* __restrict__ wv,
                  unsigned short* __restrict__ wt_)
{
  __bf16* wt = (__bf16*)wt_;
  const int mat = blockIdx.y;
  const float* w = (mat == 0) ? wq : (mat == 1 ? wk : wv);
  const int idx = blockIdx.x * 256 + threadIdx.x;   // over C_*H_
  const int c = idx >> 7;
  const int h = idx & (H_ - 1);
  wt[((size_t)mat * H_ + h) * C_ + c] = bf(w[idx]);
}

// ---------------------------------------------------------------------------
// Phase 1: one wave computes a 16x128 tile of Q, K or V (8 col-tiles share A).
// grid = (128 row-groups, 3 matrices), block = 256 (8 waves = 8 row tiles)
// ---------------------------------------------------------------------------
__global__ __launch_bounds__(256)
void qkv_rope_kernel(const float* __restrict__ x,
                     const unsigned short* __restrict__ wt_,
                     const float* __restrict__ cosb,
                     const float* __restrict__ sinb,
                     unsigned short* __restrict__ qws_,
                     unsigned short* __restrict__ kws_,
                     unsigned short* __restrict__ vws_)
{
  const __bf16* wt = (const __bf16*)wt_;
  __bf16* qws = (__bf16*)qws_;
  __bf16* kws = (__bf16*)kws_;
  __bf16* vws = (__bf16*)vws_;

  const int lane    = threadIdx.x & 31;
  const int wave    = threadIdx.x >> 5;
  const int rowTile = blockIdx.x * 8 + wave;  // 0..1023 (B*T/16 row tiles)
  const int mat     = blockIdx.y;             // 0:Q 1:K 2:V

  const int l15  = lane & 15;
  const int hi   = lane >> 4;
  const int kh8  = hi * 8;                    // C-frag row base (rows 8*hi + r)
  const int kh16 = hi * 16;                   // B-frag K base (contiguous)
  const int aRow = rowTile * 16 + l15;

  // B source rows: wT[mat][ct*16 + l15][*], contiguous in K
  const __bf16* wbase = wt + ((size_t)mat * H_ + l15) * C_;

  v8f acc[8];
  #pragma unroll
  for (int ct = 0; ct < 8; ++ct) acc[ct] = (v8f){};

  for (int kc = 0; kc < C_; kc += 32) {
    v16bf a;
    const float* xa = x + (size_t)aRow * C_ + kc + kh16;
    #pragma unroll
    for (int i = 0; i < 8; ++i) { a[i] = bf(xa[i]); a[8 + i] = bf(xa[8 + i]); }
    #pragma unroll
    for (int ct = 0; ct < 8; ++ct) {          // 8 WMMAs reuse one A fragment
      v16bf bb;
      const __bf16* wp = wbase + (size_t)(ct * 16) * C_ + kc + kh16;
      #pragma unroll
      for (int i = 0; i < 16; ++i) bb[i] = wp[i];
      acc[ct] = wmma_bf16(a, bb, acc[ct]);
    }
  }

  // C layout per tile: lane holds column ct*16+l15; VGPR r holds row r / r+8.
  if (mat < 2) {
    __bf16* dst = (mat == 0) ? qws : kws;
    #pragma unroll
    for (int ct = 0; ct < 8; ++ct) {
      const int   nCol = ct * 16 + l15;
      const float sgn  = (nCol & 1) ? 1.0f : -1.0f;
      #pragma unroll
      for (int r = 0; r < 8; ++r) {
        const int   gRow = rowTile * 16 + r + kh8;
        const int   t    = gRow & (T_ - 1);
        const float v    = acc[ct][r];
        const float vp   = __shfl_xor(v, 1);
        const float c    = cosb[t * H_ + nCol];
        const float s    = sinb[t * H_ + nCol];
        dst[(size_t)gRow * H_ + nCol] = bf(v * c + sgn * vp * s);
      }
    }
  } else {
    #pragma unroll
    for (int ct = 0; ct < 8; ++ct) {
      const int nCol = ct * 16 + l15;
      #pragma unroll
      for (int r = 0; r < 8; ++r) {
        const int gRow = rowTile * 16 + r + kh8;
        const int bb   = gRow >> 11;
        const int t    = gRow & (T_ - 1);
        vws[((size_t)bb * H_ + nCol) * T_ + t] = bf(acc[ct][r]);
      }
    }
  }
}

// ---------------------------------------------------------------------------
// Phase 2: flash attention with async K staging.
// One workgroup (8 waves) per (batch, 16-query tile); waves split key blocks.
// Dynamic LDS: per-wave double-buffered 8KB K blocks (8*2*8KB = 128KB).
// ---------------------------------------------------------------------------
__global__ __launch_bounds__(256)
void attn_kernel(const unsigned short* __restrict__ qws_,
                 const unsigned short* __restrict__ kws_,
                 const unsigned short* __restrict__ vws_,
                 float* __restrict__ out)
{
  const __bf16* qws = (const __bf16*)qws_;
  const __bf16* kws = (const __bf16*)kws_;
  const __bf16* vws = (const __bf16*)vws_;

  extern __shared__ __bf16 kstage[];   // [8 waves][2 bufs][32*128]
  __shared__ __bf16 pbuf[8][16][32];   // per-wave P transpose bounce (8 KB)
  __shared__ float  Ml[8][16];
  __shared__ float  Ll[8][16];
  __shared__ float  Osum[16][128];     // merged output tile (8 KB)
  __shared__ float  Lg[16];

  const int lane = threadIdx.x & 31;
  const int wave = threadIdx.x >> 5;   // 0..7 : key-split index
  const int b    = blockIdx.x >> 7;
  const int qt   = blockIdx.x & 127;

  const int l15  = lane & 15;
  const int hi   = lane >> 4;
  const int kh8  = hi * 8;
  const int kh16 = hi * 16;

  __bf16* kbufs = kstage + (size_t)wave * 2 * (32 * H_);

  // cooperative zero of merge buffers
  {
    const int tid = threadIdx.x;
    #pragma unroll
    for (int i = 0; i < 8; ++i) (&Osum[0][0])[tid + i * 256] = 0.0f;
    if (tid < 16) Lg[tid] = 0.0f;
  }

  // Q as 4 A-fragments
  v16bf qf[4];
  const __bf16* qrow = qws + ((size_t)(b * T_ + qt * 16 + l15)) * H_;
  #pragma unroll
  for (int kk = 0; kk < 4; ++kk) {
    const __bf16* qp = qrow + kk * 32 + kh16;
    #pragma unroll
    for (int i = 0; i < 8; ++i) { qf[kk][i] = qp[i]; qf[kk][8 + i] = qp[8 + i]; }
  }

  v8f   o[8];
  float Mrow[8], Lrow[8];
  #pragma unroll
  for (int h = 0; h < 8; ++h) o[h] = (v8f){};
  #pragma unroll
  for (int r = 0; r < 8; ++r) { Mrow[r] = -__builtin_inff(); Lrow[r] = 0.0f; }

  const float scale   = 0.08838834764831845f;       // 1/sqrt(128)
  const int   tmax    = qt * 16 + 15;
  const int   nblocks = (tmax >> 5) + 1;            // key blocks of 32

  // ---- async stage: copy one contiguous 8KB K block into this wave's LDS --
  auto stageK = [&](int bufsel, int blk) {
    const unsigned long long g =
        (unsigned long long)(uintptr_t)(kws + (size_t)(b * T_ + blk * 32) * H_) +
        (unsigned long long)(lane * 16);
    const unsigned l =
        (unsigned)(uintptr_t)(kbufs + (size_t)bufsel * (32 * H_)) +
        (unsigned)(lane * 16);
    #pragma unroll
    for (int q = 0; q < 16; ++q) {                  // 32 lanes * 16B * 16 = 8KB
      asm volatile("global_load_async_to_lds_b128 %0, %1, off"
                   :: "v"(l + (unsigned)(q * 512)),
                      "v"(g + (unsigned long long)(q * 512))
                   : "memory");
    }
  };

  int bufsel = 0;
  if (wave < nblocks) stageK(0, wave);              // prologue

  for (int blk = wave; blk < nblocks; blk += 8) {
    const int  s0      = blk * 32;
    const bool hasNext = (blk + 8) < nblocks;
    if (hasNext) {
      stageK(bufsel ^ 1, blk + 8);                  // overlap: next block
      __builtin_prefetch(vws + ((size_t)b * H_ + lane * 4) * T_ + (blk + 8) * 32, 0, 1);
      asm volatile("s_wait_asynccnt 16" ::: "memory");   // retire current buffer only
    } else {
      asm volatile("s_wait_asynccnt 0" ::: "memory");
    }
    const __bf16* kblk = kbufs + (size_t)bufsel * (32 * H_);   // 32 keys x 128

    // ---- S(16x32) = Q * K^T : two 16x16 tiles (K from LDS) ---------------
    v8f S[2];
    #pragma unroll
    for (int j = 0; j < 2; ++j) {
      S[j] = (v8f){};
      const __bf16* krow = kblk + (size_t)(j * 16 + l15) * H_;
      #pragma unroll
      for (int kk = 0; kk < 4; ++kk) {
        v16bf kb;
        const __bf16* kp = krow + kk * 32 + kh16;
        #pragma unroll
        for (int i = 0; i < 16; ++i) kb[i] = kp[i];
        S[j] = wmma_bf16(qf[kk], kb, S[j]);
      }
    }

    // ---- scale + causal mask + online softmax ----------------------------
    #pragma unroll
    for (int r = 0; r < 8; ++r) {
      const int tq = qt * 16 + r + kh8;
      #pragma unroll
      for (int j = 0; j < 2; ++j) {
        const int key = s0 + j * 16 + l15;
        float sv = S[j][r] * scale;
        if (key > tq) sv = -__builtin_inff();
        S[j][r] = sv;
      }
      float mx = fmaxf(S[0][r], S[1][r]);
      #pragma unroll
      for (int d = 1; d < 16; d <<= 1) mx = fmaxf(mx, __shfl_xor(mx, d));
      const float Mn   = fmaxf(Mrow[r], mx);
      const float corr = __expf(Mrow[r] - Mn);
      Lrow[r] *= corr;
      #pragma unroll
      for (int h = 0; h < 8; ++h) o[h][r] *= corr;
      float psum = 0.0f;
      #pragma unroll
      for (int j = 0; j < 2; ++j) {
        const float p = __expf(S[j][r] - Mn);
        S[j][r] = p;
        psum += p;
      }
      #pragma unroll
      for (int d = 1; d < 16; d <<= 1) psum += __shfl_xor(psum, d);
      Lrow[r] += psum;
      Mrow[r]  = Mn;
    }

    // ---- C-layout P -> A-layout fragment via per-wave LDS bounce ----------
    #pragma unroll
    for (int r = 0; r < 8; ++r)
      #pragma unroll
      for (int j = 0; j < 2; ++j)
        pbuf[wave][r + kh8][j * 16 + l15] = bf(S[j][r]);
    asm volatile("s_wait_dscnt 0" ::: "memory");
    v16bf pf;
    {
      const __bf16* pr = &pbuf[wave][l15][0];
      #pragma unroll
      for (int i = 0; i < 8; ++i) { pf[i] = pr[kh16 + i]; pf[8 + i] = pr[kh16 + 8 + i]; }
    }

    // ---- O(16x128) += P(16x32) * V(32x128) -------------------------------
    #pragma unroll
    for (int ht = 0; ht < 8; ++ht) {
      v16bf vb;
      const __bf16* vp = vws + ((size_t)b * H_ + ht * 16 + l15) * T_ + s0 + kh16;
      #pragma unroll
      for (int i = 0; i < 16; ++i) vb[i] = vp[i];
      o[ht] = wmma_bf16(pf, vb, o[ht]);
    }

    bufsel ^= 1;
  }

  // ---- publish per-wave softmax stats ------------------------------------
  if (l15 == 0) {
    #pragma unroll
    for (int r = 0; r < 8; ++r) {
      Ml[wave][kh8 + r] = Mrow[r];
      Ll[wave][kh8 + r] = Lrow[r];
    }
  }
  __syncthreads();

  // ---- merge: rescale by exp(M_w - M_g), accumulate via LDS atomics -------
  #pragma unroll
  for (int r = 0; r < 8; ++r) {
    const int row = kh8 + r;
    float mg = Ml[0][row];
    #pragma unroll
    for (int w = 1; w < 8; ++w) mg = fmaxf(mg, Ml[w][row]);
    const float f = __expf(Mrow[r] - mg);           // 0 for idle/empty waves
    if (l15 == 0) atomicAdd(&Lg[row], Lrow[r] * f);
    #pragma unroll
    for (int ht = 0; ht < 8; ++ht)
      atomicAdd(&Osum[row][ht * 16 + l15], o[ht][r] * f);
  }
  __syncthreads();

  // ---- normalized write-out (coalesced) ----------------------------------
  {
    const int tid = threadIdx.x;
    const int col = tid & 127;
    const int rb  = (tid >> 7) * 8;
    #pragma unroll
    for (int rr = 0; rr < 8; ++rr) {
      const int row = rb + rr;
      const int tq  = qt * 16 + row;
      out[((size_t)b * T_ + tq) * H_ + col] = Osum[row][col] / Lg[row];
    }
  }
}

// ---------------------------------------------------------------------------
extern "C" void kernel_launch(void* const* d_in, const int* in_sizes, int n_in,
                              void* d_out, int out_size, void* d_ws, size_t ws_size,
                              hipStream_t stream) {
  const float* x  = (const float*)d_in[0];
  const float* wq = (const float*)d_in[1];
  const float* wk = (const float*)d_in[2];
  const float* wv = (const float*)d_in[3];
  const float* rc = (const float*)d_in[4];
  const float* rs = (const float*)d_in[5];

  const size_t qkvBytes = (size_t)B_ * T_ * H_ * 2;   // 4 MiB each (bf16)
  char* ws = (char*)d_ws;
  unsigned short* qws = (unsigned short*)(ws);
  unsigned short* kws = (unsigned short*)(ws + qkvBytes);
  unsigned short* vws = (unsigned short*)(ws + 2 * qkvBytes);
  unsigned short* wt  = (unsigned short*)(ws + 3 * qkvBytes);  // 1.5 MiB

  wconv_kernel<<<dim3(C_ * H_ / 256, 3), dim3(256), 0, stream>>>(wq, wk, wv, wt);

  qkv_rope_kernel<<<dim3(B_ * T_ / 16 / 8, 3), dim3(256), 0, stream>>>(
      x, wt, rc, rs, qws, kws, vws);

  const size_t kstageBytes = (size_t)8 * 2 * 32 * H_ * 2;      // 128 KiB dynamic LDS
  attn_kernel<<<dim3(B_ * (T_ / 16)), dim3(256), kstageBytes, stream>>>(
      qws, kws, vws, (float*)d_out);
}
